// LSTM_decay_48326972015011
// MI455X (gfx1250) — matrix operand order repaired and checked by reading, not compiled, not run on hardware
//
#include <hip/hip_runtime.h>
#include <hip/hip_bf16.h>

typedef __attribute__((ext_vector_type(16))) __bf16 v16bf;
typedef __attribute__((ext_vector_type(8)))  __bf16 v8bf;
typedef __attribute__((ext_vector_type(8)))  float  v8f;
typedef __attribute__((ext_vector_type(4)))  float  v4f;
typedef __attribute__((ext_vector_type(4)))  unsigned int v4ui;
typedef __attribute__((ext_vector_type(8)))  int    v8i;
typedef __attribute__((ext_vector_type(4)))  int    v4i;

#define Bdim 4096
#define Tdim 48
#define INd  128
#define Hd   512
#define MT   32           // batch rows per workgroup
#define NWAVES 8
#define CTW  4            // column tiles (of 16) per wave: 32 tiles / 8 waves
#define KT_IN (INd/32)    // 4
#define KT_H  (Hd/32)     // 16

#if __has_builtin(__builtin_amdgcn_tensor_load_to_lds)
#define USE_TDM 1
#else
#define USE_TDM 0
#endif

// ---------------------------------------------------------------------------
// Pack fp32 weight [K,N] (optionally stored transposed) into bf16 B-operand
// blocks. Block (nt,kt) = 512 bf16: lane l (0..31) holds 16 contiguous bf16:
//   lane<16 : W[kt*32 +  0..15][nt*16+lane]
//   lane>=16: W[kt*32 + 16..31][nt*16+lane-16]
// ---------------------------------------------------------------------------
__global__ void pack_b_kernel(const float* __restrict__ src, __bf16* __restrict__ dst,
                              int K, int N, int ld, int transposed) {
    int p = blockIdx.x * 256 + threadIdx.x;
    if (p >= K * N) return;
    int ktiles = K >> 5;
    int blk  = p >> 9;
    int r    = p & 511;
    int lane = r >> 4;
    int j    = r & 15;
    int kt = blk % ktiles;
    int nt = blk / ktiles;
    int k = (kt << 5) + ((lane >> 4) << 4) + j;
    int n = (nt << 4) + (lane & 15);
    float v = transposed ? src[(size_t)n * ld + k] : src[(size_t)k * ld + n];
    dst[p] = (__bf16)v;
}

__device__ __forceinline__ v16bf cat16(v8bf lo, v8bf hi) {
    return __builtin_shufflevector(lo, hi, 0,1,2,3,4,5,6,7,8,9,10,11,12,13,14,15);
}

template<int S>
__device__ __forceinline__ v16bf load_a(const __bf16* base, int lane, int m0, int kb) {
    const __bf16* p = base + (size_t)(m0 + (lane & 15)) * S + kb + ((lane >> 4) << 4);
    v8bf lo = *(const v8bf*)p;
    v8bf hi = *(const v8bf*)(p + 8);
    return cat16(lo, hi);
}

__device__ __forceinline__ v16bf load_b(const __bf16* blk, int lane) {
    const __bf16* p = blk + lane * 16;
    v8bf lo = *(const v8bf*)p;
    v8bf hi = *(const v8bf*)(p + 8);
    return cat16(lo, hi);
}

#if USE_TDM
// LDS byte offset of a __shared__ object: flat LDS address truncates to offset
// (ISA 10.2).
__device__ __forceinline__ unsigned lds_off(const void* p) {
    return (unsigned)(unsigned long long)p;
}

// Issue a TDM 2D tile load: rows x dim0 fp32 elements, row stride strideE
// elements, global -> LDS at ldsOff. D# per CDNA5 ISA §8.3/8.4.
__device__ __forceinline__ void tdm_load_2d(unsigned ldsOff, const float* gptr,
                                            unsigned dim0, unsigned rows,
                                            unsigned strideE) {
    unsigned long long ga = (unsigned long long)gptr;
    v4ui g0;
    g0[0] = 1u;                                            // count=1, user mode
    g0[1] = ldsOff;                                        // lds_addr
    g0[2] = (unsigned)(ga & 0xFFFFFFFFu);                  // global_addr[31:0]
    g0[3] = (unsigned)((ga >> 32) & 0x01FFFFFFu) | (2u << 30); // addr[56:32]|type=2
    v8i g1;
    g1[0] = (int)(2u << 16);                               // data_size = 4 bytes
    g1[1] = (int)((dim0 & 0xFFFFu) << 16);                 // tensor_dim0[15:0]
    g1[2] = (int)(((dim0 >> 16) & 0xFFFFu) | ((rows & 0xFFFFu) << 16)); // dim0 hi | tensor_dim1 lo
    g1[3] = (int)(((rows >> 16) & 0xFFFFu) | ((dim0 & 0xFFFFu) << 16)); // dim1 hi | tile_dim0
    g1[4] = (int)(rows & 0xFFFFu);                         // tile_dim1 (tile_dim2=0)
    g1[5] = (int)strideE;                                  // tensor_dim0_stride[31:0]
    g1[6] = 0;                                             // stride0 hi | stride1 lo
    g1[7] = 0;                                             // stride1 hi
    v4i z4 = (v4i){0, 0, 0, 0};
#if __clang_major__ >= 23
    v8i z8 = (v8i){0, 0, 0, 0, 0, 0, 0, 0};
    __builtin_amdgcn_tensor_load_to_lds(g0, g1, z4, z4, z8, 0);
#else
    __builtin_amdgcn_tensor_load_to_lds(g0, g1, z4, z4, 0);
#endif
}
#endif // USE_TDM

// ---------------------------------------------------------------------------
// Persistent-per-timestep recurrence kernel. One WG = 32 batch rows, all T.
// ---------------------------------------------------------------------------
__global__ __launch_bounds__(256, 1)
void grud_fused_kernel(const float* __restrict__ X, const float* __restrict__ Mk,
                       const float* __restrict__ b_td, const float* __restrict__ b_h,
                       const __bf16* __restrict__ Wtd_p, const __bf16* __restrict__ Wh_p,
                       const __bf16* __restrict__ Uh_p, float* __restrict__ out)
{
    __shared__ float  h32[MT][Hd];     // fp32 recurrent state
    __shared__ __bf16 hbf[MT][Hd];     // bf16 A-staging of h*gamma
    __shared__ __bf16 dbf[MT][INd];    // decay counters (bf16-exact, <=48)
    __shared__ __bf16 xmbf[MT][INd];   // m*x
#if USE_TDM
    __shared__ float  xstg[2][MT][INd]; // double-buffered TDM landing pads
    __shared__ float  mstg[2][MT][INd];
#endif

    const int tid  = threadIdx.x;
    const int lane = tid & 31;
    const int wave = tid >> 5;
    const int row0 = blockIdx.x * MT;

#if USE_TDM
    // Prime the pipeline: DMA step-0 tiles (wave0: X, wave1: M). TDM ignores
    // EXEC but is issued per-wave, so guard by wave to avoid duplicate DMAs.
    {
        const float* xg = X  + ((size_t)row0 * Tdim + 0) * INd;
        const float* mg = Mk + ((size_t)row0 * Tdim + 0) * INd;
        if (wave == 0)      tdm_load_2d(lds_off(&xstg[0][0][0]), xg, INd, MT, Tdim * INd);
        else if (wave == 1) tdm_load_2d(lds_off(&mstg[0][0][0]), mg, INd, MT, Tdim * INd);
    }
#endif

    for (int i = tid; i < MT * Hd; i += 256) (&h32[0][0])[i] = 0.0f;

    // each thread owns 16 (row,feature) decay elements
    const int drow = tid >> 3;
    const int f0   = (tid & 7) << 4;
    float dreg[16];
    #pragma unroll
    for (int j = 0; j < 16; ++j) dreg[j] = 0.0f;

    // per-lane bias values for this wave's 4 column tiles (hoisted out of t-loop)
    float btd[CTW], bhv[CTW];
    #pragma unroll
    for (int ct = 0; ct < CTW; ++ct) {
        int n = ((wave * CTW + ct) << 4) + (lane & 15);
        btd[ct] = b_td[n];
        bhv[ct] = b_h[n];
    }

    __syncthreads();

    for (int t = 0; t < Tdim; ++t) {
#if USE_TDM
        // ---- async prefetch of step t+1; wait for step t's DMA to land ----
        if (t + 1 < Tdim) {
            const int nb = (t + 1) & 1;
            const float* xg = X  + ((size_t)row0 * Tdim + (t + 1)) * INd;
            const float* mg = Mk + ((size_t)row0 * Tdim + (t + 1)) * INd;
            if (wave == 0)      tdm_load_2d(lds_off(&xstg[nb][0][0]), xg, INd, MT, Tdim * INd);
            else if (wave == 1) tdm_load_2d(lds_off(&mstg[nb][0][0]), mg, INd, MT, Tdim * INd);
            if (wave < 2) __builtin_amdgcn_s_wait_tensorcnt(1); // step-t DMA done (in-order)
        } else {
            if (wave < 2) __builtin_amdgcn_s_wait_tensorcnt(0);
        }
        __syncthreads();   // publish DMA'd tiles to all waves
#endif

        // ---- phase 0: stage d (current) and m*x; advance d recurrence ----
        {
#if !USE_TDM
            const size_t gbase = ((size_t)(row0 + drow) * Tdim + t) * INd + f0;
#endif
            v8bf dv0, dv1, xv0, xv1;
            #pragma unroll
            for (int q = 0; q < 4; ++q) {
#if USE_TDM
                v4f xv = *(const v4f*)&xstg[t & 1][drow][f0 + q * 4];
                v4f mv = *(const v4f*)&mstg[t & 1][drow][f0 + q * 4];
#else
                v4f xv = *(const v4f*)(X  + gbase + q * 4);
                v4f mv = *(const v4f*)(Mk + gbase + q * 4);
#endif
                #pragma unroll
                for (int e = 0; e < 4; ++e) {
                    int j = q * 4 + e;
                    __bf16 db = (__bf16)dreg[j];
                    __bf16 xb = (__bf16)(xv[e] * mv[e]);
                    if (j < 8) { dv0[j] = db; xv0[j] = xb; }
                    else       { dv1[j - 8] = db; xv1[j - 8] = xb; }
                    dreg[j] = 1.0f + (1.0f - mv[e]) * dreg[j];
                }
            }
            *(v8bf*)&dbf[drow][f0]      = dv0;
            *(v8bf*)&dbf[drow][f0 + 8]  = dv1;
            *(v8bf*)&xmbf[drow][f0]     = xv0;
            *(v8bf*)&xmbf[drow][f0 + 8] = xv1;
        }
        __syncthreads();

        // ---- phase 1: gamma = exp(-relu(d @ WtdT + b_td)); hbf = h32*gamma ----
        #pragma unroll
        for (int rt = 0; rt < 2; ++rt) {
            v8f acc[CTW];
            #pragma unroll
            for (int ct = 0; ct < CTW; ++ct) acc[ct] = (v8f){0,0,0,0,0,0,0,0};
            #pragma unroll
            for (int kt = 0; kt < KT_IN; ++kt) {
                v16bf a = load_a<INd>(&dbf[0][0], lane, rt * 16, kt * 32);
                #pragma unroll
                for (int ct = 0; ct < CTW; ++ct) {
                    int ntg = wave * CTW + ct;
                    v16bf b = load_b(Wtd_p + ((size_t)(ntg * KT_IN + kt) << 9), lane);
                    acc[ct] = __builtin_amdgcn_wmma_f32_16x16x32_bf16(
                        false, a, false, b, (short)0, acc[ct], false, false);
                }
            }
            #pragma unroll
            for (int ct = 0; ct < CTW; ++ct) {
                int n = ((wave * CTW + ct) << 4) + (lane & 15);
                #pragma unroll
                for (int v = 0; v < 8; ++v) {
                    int m = rt * 16 + v + ((lane >> 4) << 3);
                    float g  = __expf(-fmaxf(acc[ct][v] + btd[ct], 0.0f));
                    hbf[m][n] = (__bf16)(h32[m][n] * g);
                }
            }
        }
        __syncthreads();

        // ---- phase 2: h = sigmoid(hbf@W_h + xm@U_h + b_h); store out ----
        #pragma unroll
        for (int rt = 0; rt < 2; ++rt) {
            v8f acc[CTW];
            #pragma unroll
            for (int ct = 0; ct < CTW; ++ct) acc[ct] = (v8f){0,0,0,0,0,0,0,0};
            #pragma unroll 4
            for (int kt = 0; kt < KT_H; ++kt) {
                v16bf a = load_a<Hd>(&hbf[0][0], lane, rt * 16, kt * 32);
                #pragma unroll
                for (int ct = 0; ct < CTW; ++ct) {
                    int ntg = wave * CTW + ct;
                    v16bf b = load_b(Wh_p + ((size_t)(ntg * KT_H + kt) << 9), lane);
                    acc[ct] = __builtin_amdgcn_wmma_f32_16x16x32_bf16(
                        false, a, false, b, (short)0, acc[ct], false, false);
                }
            }
            #pragma unroll
            for (int kt = 0; kt < KT_IN; ++kt) {
                v16bf a = load_a<INd>(&xmbf[0][0], lane, rt * 16, kt * 32);
                #pragma unroll
                for (int ct = 0; ct < CTW; ++ct) {
                    int ntg = wave * CTW + ct;
                    v16bf b = load_b(Uh_p + ((size_t)(ntg * KT_IN + kt) << 9), lane);
                    acc[ct] = __builtin_amdgcn_wmma_f32_16x16x32_bf16(
                        false, a, false, b, (short)0, acc[ct], false, false);
                }
            }
            #pragma unroll
            for (int ct = 0; ct < CTW; ++ct) {
                int n = ((wave * CTW + ct) << 4) + (lane & 15);
                #pragma unroll
                for (int v = 0; v < 8; ++v) {
                    int m = rt * 16 + v + ((lane >> 4) << 3);
                    float z = acc[ct][v] + bhv[ct];
                    float hnew = 1.0f / (1.0f + __expf(-z));
                    h32[m][n] = hnew;
                    // H (403MB/call) > L2 (192MB): stream out non-temporally so
                    // weights + X/M stay cache-resident.
                    __builtin_nontemporal_store(
                        hnew, &out[((size_t)(row0 + m) * Tdim + t) * Hd + n]);
                }
            }
        }
        __syncthreads();
    }
}

extern "C" void kernel_launch(void* const* d_in, const int* in_sizes, int n_in,
                              void* d_out, int out_size, void* d_ws, size_t ws_size,
                              hipStream_t stream) {
    (void)in_sizes; (void)n_in; (void)out_size; (void)ws_size;
    const float* X    = (const float*)d_in[0];
    const float* Mk   = (const float*)d_in[1];
    const float* W_td = (const float*)d_in[2];
    const float* b_td = (const float*)d_in[3];
    // d_in[4] = W_hr, d_in[5] = b_hr: provably dead for binary mask (x_c*m == m*x)
    const float* W_h  = (const float*)d_in[6];
    const float* U_h  = (const float*)d_in[7];
    const float* b_h  = (const float*)d_in[8];
    float* out = (float*)d_out;

    __bf16* Wtd_p = (__bf16*)d_ws;              // [128,512] packed  (W_td transposed)
    __bf16* Wh_p  = Wtd_p + INd * Hd;           // [512,512] packed
    __bf16* Uh_p  = Wh_p  + Hd * Hd;            // [128,512] packed

    pack_b_kernel<<<(INd * Hd + 255) / 256, 256, 0, stream>>>(W_td, Wtd_p, INd, Hd, INd, 1);
    pack_b_kernel<<<(Hd  * Hd + 255) / 256, 256, 0, stream>>>(W_h,  Wh_p,  Hd,  Hd, Hd, 0);
    pack_b_kernel<<<(INd * Hd + 255) / 256, 256, 0, stream>>>(U_h,  Uh_p,  INd, Hd, Hd, 0);

    grud_fused_kernel<<<Bdim / MT, 256, 0, stream>>>(X, Mk, b_td, b_h,
                                                     Wtd_p, Wh_p, Uh_p, out);
}
